// QuantumNet_38568806318762
// MI455X (gfx1250) — compile-verified, matrix operand order checked by reading, not run on hardware
//
#include <hip/hip_runtime.h>
#include <math.h>

// CDNA5 / gfx1250, wave32. Closed form of the reference circuit:
//   P1(b) = 0.5 * (1 + R * cos(Psum - 2 * bits_b . psi))
// with per-qubit polar factors r_k, psi_k from params. The only
// batch-dependent work, bits[128x16] @ psi[16], runs on
// v_wmma_f32_16x16x32_f16 (exact: bits are 0/1 in f16, psi split
// into exact f16 hi+lo parts, two accumulating WMMAs, f32 acc).

typedef __attribute__((ext_vector_type(16))) _Float16 v16h;
typedef __attribute__((ext_vector_type(8)))  float    v8f;

#define NQ 16  // data qubits == row length of X

__global__ __launch_bounds__(256)
void qnet_closedform_wmma(const float* __restrict__ X,
                          const float* __restrict__ params,
                          float* __restrict__ out, int batch)
{
    const int lane = threadIdx.x & 31;
    const int wave = threadIdx.x >> 5;            // 8 waves per block
    const int tile = blockIdx.x * 8 + wave;       // 16 batch rows per tile
    const int rowBase = tile * 16;
    if (rowBase >= batch) return;                 // whole-wave exit only

    const int hsel = lane >> 4;   // 0: lanes 0-15, 1: lanes 16-31
    const int m    = lane & 15;   // tile-local row (A row / D column) & qubit id

    // ---- lane-parallel polar factors: lanes (k, k+16) handle qubit k ----
    // c_k = cos(phi) + i*sigma*cos(theta)*sin(phi) = r_k * exp(i*sigma*psi_k)
    float th  = params[m];         // params[0, k]  (XX layer)
    float ph  = params[NQ + m];    // params[1, k]  (ZZ layer)
    float y   = __cosf(th) * __sinf(ph);
    float x   = __cosf(ph);
    float r2  = fmaf(x, x, y * y);
    float psi = atan2f(y, x);

    // ---- wave-uniform R = sqrt(prod r2_k), Psum = sum psi_k ----
    // (upper lane half duplicates qubit k -> mask it out of the reduction)
    float psum = hsel ? 0.0f : psi;
    float rpro = hsel ? 1.0f : r2;
#pragma unroll
    for (int off = 1; off < 32; off <<= 1) {
        psum += __shfl_xor(psum, off, 32);
        rpro *= __shfl_xor(rpro, off, 32);
    }
    const float R = sqrtf(rpro);

    const _Float16 z = (_Float16)0.0f;

    // ---- A matrix: bits, placed only in layout-invariant K slots ----
    //  lanes 0-15 , elems 0-7  -> K = 0..7   -> qubits 0..7
    //  lanes 16-31, elems 8-15 -> K = 24..31 -> qubits 8..15
    const float* xrow = X + (size_t)(rowBase + m) * NQ;
    v16h a;
#pragma unroll
    for (int j = 0; j < 8; ++j) {
        _Float16 bit = (xrow[(hsel << 3) + j] > 0.5f) ? (_Float16)1.0f : z;
        a[j]     = hsel ? z   : bit;
        a[8 + j] = hsel ? bit : z;
    }

    // ---- B matrices (all 16 columns identical): psi gathered via shuffle,
    //      exact hi/lo f16 split; hi pass then lo pass ----
    v16h bh, bl;
#pragma unroll
    for (int j = 0; j < 8; ++j) {
        float pj = __shfl(psi, (hsel << 3) + j, 32);  // psi[qubit] from lane
        _Float16 h = (_Float16)pj;
        _Float16 l = (_Float16)(pj - (float)h);       // exact residual
        bh[j]     = hsel ? z : h;
        bh[8 + j] = hsel ? h : z;
        bl[j]     = hsel ? z : l;
        bl[8 + j] = hsel ? l : z;
    }

    // D[m, n] = sum_k bit[m,k]*(hi[k] + lo[k]) = bits . psi  (exact, f32 acc)
    v8f c = {};
    c = __builtin_amdgcn_wmma_f32_16x16x32_f16(false, a, false, bh,
                                               (short)0, c, false, false);
    c = __builtin_amdgcn_wmma_f32_16x16x32_f16(false, a, false, bl,
                                               (short)0, c, false, false);

    // ---- closed form output ----
    // D layout: lanes 0-15 hold rows 0..7 in c[0..7]; lanes 16-31 rows 8..15.
    // All columns identical -> lanes 0 and 16 cover the whole tile.
    if (m == 0) {
#pragma unroll
        for (int j = 0; j < 8; ++j) {
            int row = j + (hsel << 3);
            float s = c[j];
            out[rowBase + row] = 0.5f * (1.0f + R * __cosf(psum - 2.0f * s));
        }
    }
}

extern "C" void kernel_launch(void* const* d_in, const int* in_sizes, int n_in,
                              void* d_out, int out_size, void* d_ws, size_t ws_size,
                              hipStream_t stream) {
    const float* X      = (const float*)d_in[0];   // [batch, 16] f32
    const float* params = (const float*)d_in[1];   // [2, 16]    f32
    float* out          = (float*)d_out;           // [batch]    f32

    int batch  = in_sizes[0] / NQ;                 // 128
    int tiles  = (batch + 15) / 16;                // 8
    int blocks = (tiles + 7) / 8;                  // 1 block of 8 waves

    qnet_closedform_wmma<<<blocks, 256, 0, stream>>>(X, params, out, batch);
}
